// ArrowTokenLM_67370857005405
// MI455X (gfx1250) — matrix operand order repaired and checked by reading, not compiled, not run on hardware
//
#include <hip/hip_runtime.h>
#include <hip/hip_bf16.h>
#include <stdint.h>

typedef __bf16 bf16_t;
typedef __attribute__((ext_vector_type(16))) __bf16 v16bf;
typedef __attribute__((ext_vector_type(4)))  __bf16 v4bf;
typedef __attribute__((ext_vector_type(8)))  float  v8f;
typedef __attribute__((ext_vector_type(4)))  unsigned uv4;
typedef __attribute__((ext_vector_type(8)))  int      iv8;
typedef __attribute__((ext_vector_type(4)))  int      iv4;

#define Vv   32000
#define Dd   1024
#define Bb   4
#define Tt   512
#define Mm   (Bb*Tt)   // 2048

#define BM 128
#define BN 128
#define BK 32
#define KPAD 8         // LDS row pad: 8 bf16 = 4 DWORDs

// ---------------- init: zero the grid-barrier counter, broadcast h0 ----------------
__global__ void atlm_init(const float* __restrict__ h0, float* __restrict__ hA,
                          unsigned* __restrict__ cnt) {
  int tid = blockIdx.x * blockDim.x + threadIdx.x;
  if (tid == 0) *cnt = 0u;
  if (tid < Bb * Dd) hA[tid] = h0[tid & (Dd - 1)];
}

// ---------------- embedding gather: e[bt,:] = emb[x[bt],:] ----------------
__global__ void atlm_gather(const int* __restrict__ x, const float* __restrict__ emb,
                            float* __restrict__ e) {
  int bt = blockIdx.x;
  int tok = x[bt];
  const float4* src = (const float4*)(emb + (size_t)tok * Dd);
  float4* dst = (float4*)(e + (size_t)bt * Dd);
  dst[threadIdx.x] = src[threadIdx.x];   // 256 threads * float4 = 1024 floats
}

// ---------------- fp32 -> bf16 weight conversion (out_w stays L2-resident) ----------------
__global__ void atlm_cvt_w(const float* __restrict__ w, bf16_t* __restrict__ wb) {
  size_t idx = ((size_t)blockIdx.x * 256 + threadIdx.x) * 4;
  float4 f = *(const float4*)(w + idx);
  v4bf o = { (__bf16)f.x, (__bf16)f.y, (__bf16)f.z, (__bf16)f.w };
  *(v4bf*)(wb + idx) = o;
}

// ---------------- persistent cooperative RNN scan ----------------
__global__ void __launch_bounds__(256) atlm_rnn(
    const float* __restrict__ e, const float* __restrict__ Uw,
    float* __restrict__ hA, float* __restrict__ hB,
    bf16_t* __restrict__ hs, unsigned* __restrict__ cnt) {
  __shared__ float Uld[8][Dd];
  const int tid = threadIdx.x;
  const int jbase = blockIdx.x * 8;
  for (int i = tid; i < 8 * Dd; i += 256) {
    int jj = i >> 10, k = i & (Dd - 1);
    Uld[jj][k] = Uw[(size_t)(jbase + jj) * Dd + k];
  }
  __syncthreads();

  const int pair = tid >> 3, lane8 = tid & 7;
  const int b = pair >> 3, jj = pair & 7;
  const int jg = jbase + jj;
  const int kstart = lane8 * 128;
  const unsigned nwg = gridDim.x;

  for (int t = 0; t < Tt; ++t) {
    const float* hc = (t & 1) ? hB : hA;
    float*       hn = (t & 1) ? hA : hB;
    float s = 0.f;
    const float4* hv = (const float4*)(hc + b * Dd + kstart);
    const float4* uv = (const float4*)(&Uld[jj][kstart]);
    #pragma unroll 8
    for (int k = 0; k < 32; ++k) {
      float4 h4 = hv[k], u4 = uv[k];
      s += h4.x * u4.x + h4.y * u4.y + h4.z * u4.z + h4.w * u4.w;
    }
    s += __shfl_down(s, 4, 8);
    s += __shfl_down(s, 2, 8);
    s += __shfl_down(s, 1, 8);
    if (lane8 == 0) {
      size_t row = (size_t)(b * Tt + t) * Dd;
      float v = tanhf(e[row + jg] + s);
      hn[b * Dd + jg] = v;
      hs[row + jg] = (bf16_t)v;         // bf16 activations for the WMMA GEMM
    }
    // -------- grid-wide barrier (monotonic counter; re-zeroed each launch) --------
    __syncthreads();
    if (tid == 0) {
      __threadfence();
      atomicAdd(cnt, 1u);
      unsigned target = nwg * (unsigned)(t + 1);
      while (__hip_atomic_load(cnt, __ATOMIC_ACQUIRE, __HIP_MEMORY_SCOPE_AGENT) < target)
        __builtin_amdgcn_s_sleep(1);
      __threadfence();
    }
    __syncthreads();
  }
}

// ---------------- TDM: 2D tile (128 rows x 32 bf16) global -> padded LDS ----------------
// D# per CDNA5 ISA ch.8: group0 = {flags, lds_addr, global_addr, type},
// group1 = {data_size/pad, tensor dims, tile dims, strides}. Pad: every 16 DWORDs
// (one 64B row) insert 4 DWORDs -> matches LDS row stride of BK+KPAD bf16.
__device__ __forceinline__ void tdm_load_tile(unsigned lds_addr, const bf16_t* gptr,
                                              unsigned tensor_rows) {
  unsigned long long ga = (unsigned long long)(uintptr_t)gptr;
  uv4 g0 = {};
  g0[0] = 1u;                                        // count=1, user descriptor
  g0[1] = lds_addr;                                  // lds byte address
  g0[2] = (unsigned)(ga & 0xFFFFFFFFu);              // global_addr[31:0]
  g0[3] = (unsigned)((ga >> 32) & 0x01FFFFFFu)       // global_addr[56:32]
        | (2u << 30);                                // type = 2 ("image")
  iv8 g1 = {};
  g1[0] = (1 << 16)                                  // data_size = 2 bytes
        | (1 << 20)                                  // pad_enable
        | (3 << 22)                                  // pad_interval: 16 DWORDs
        | (3 << 25);                                 // pad_amount: 4 DWORDs
  g1[1] = (int)((unsigned)Dd << 16);                 // tensor_dim0 = 1024 (lo16)
  g1[2] = (int)(((unsigned)Dd >> 16)                 // tensor_dim0 (hi16)
        | ((tensor_rows & 0xFFFFu) << 16));          // tensor_dim1 (lo16)
  g1[3] = (int)(((tensor_rows >> 16) & 0xFFFFu)      // tensor_dim1 (hi16)
        | ((unsigned)BK << 16));                     // tile_dim0 = 32 elems
  g1[4] = BM;                                        // tile_dim1 = 128, tile_dim2 = 0
  g1[5] = Dd;                                        // tensor_dim0_stride = 1024 (lo32)
  g1[6] = 0;                                         // stride hi / dim1_stride (unused 2D)
  g1[7] = 0;
  iv4 z4 = {};
  iv8 z8 = {};
  __builtin_amdgcn_tensor_load_to_lds(g0, g1, z4, z4, z8, 0);  // 6-arg (clang-23) form
}

// ---------------- bf16 WMMA output projection: [2048,1024] x [32000,1024]^T ----------------
union FragU { uint4 u[2]; v16bf v; };

__global__ void __launch_bounds__(256) atlm_gemm(
    const bf16_t* __restrict__ A,   // hs  [M,K] row-major
    const bf16_t* __restrict__ Bw,  // out_w bf16 [N,K] row-major (== B col-major)
    const float* __restrict__ bias, float* __restrict__ C) {
  __shared__ bf16_t As[2][BM][BK + KPAD];   // double-buffered, TDM-padded rows
  __shared__ bf16_t Bs[2][BN][BK + KPAD];
  const int tid  = threadIdx.x;
  const int gN   = blockIdx.x * BN;
  const int gM   = blockIdx.y * BM;
  const int wave = tid >> 5, lane = tid & 31;
  const int wm = wave >> 2, wn = wave & 3;       // 2x4 wave grid
  const int half = lane >> 4, l16 = lane & 15;

  const bf16_t* aTile = A  + (size_t)gM * Dd;
  const bf16_t* bTile = Bw + (size_t)gN * Dd;
  const unsigned ldsA[2] = { (unsigned)(uintptr_t)&As[0][0][0],
                             (unsigned)(uintptr_t)&As[1][0][0] };
  const unsigned ldsB[2] = { (unsigned)(uintptr_t)&Bs[0][0][0],
                             (unsigned)(uintptr_t)&Bs[1][0][0] };

  v8f zero = {};
  v8f acc[4][2];
  #pragma unroll
  for (int i = 0; i < 4; ++i)
    #pragma unroll
    for (int j = 0; j < 2; ++j) acc[i][j] = zero;

  const int nK = Dd / BK;                        // 32 K-chunks
  if (wave == 0) {                               // one wave drives the TDM
    tdm_load_tile(ldsA[0], aTile, Mm);
    tdm_load_tile(ldsB[0], bTile, Vv);
  }

  for (int i = 0; i < nK; ++i) {
    const int cur = i & 1;
    if (wave == 0) {
      if (i + 1 < nK) {                          // prefetch tiles i+1 into other buffer
        tdm_load_tile(ldsA[cur ^ 1], aTile + (i + 1) * BK, Mm);
        tdm_load_tile(ldsB[cur ^ 1], bTile + (i + 1) * BK, Vv);
        __builtin_amdgcn_s_wait_tensorcnt((short)2);  // in-order: tiles i landed
      } else {
        __builtin_amdgcn_s_wait_tensorcnt((short)0);
      }
    }
    __syncthreads();                             // tile i visible to all waves

    // A frag 16x32 bf16: lane m=l16; lanes<16 -> K {0..7,16..23}, lanes>=16 -> {8..15,24..31}
    v16bf afrag[4], bfrag[2];
    #pragma unroll
    for (int mi = 0; mi < 4; ++mi) {
      int m = wm * 64 + mi * 16 + l16;
      FragU f;
      f.u[0] = *(const uint4*)&As[cur][m][half * 8];
      f.u[1] = *(const uint4*)&As[cur][m][half * 8 + 16];
      afrag[mi] = f.v;
    }
    // B frag 32x16 bf16: lane n=l16; lanes<16 -> K {0..7,16..23}, lanes>=16 -> {8..15,24..31}
    #pragma unroll
    for (int ni = 0; ni < 2; ++ni) {
      int n = wn * 32 + ni * 16 + l16;
      FragU f;
      f.u[0] = *(const uint4*)&Bs[cur][n][half * 8];
      f.u[1] = *(const uint4*)&Bs[cur][n][half * 8 + 16];
      bfrag[ni] = f.v;
    }
    #pragma unroll
    for (int mi = 0; mi < 4; ++mi)
      #pragma unroll
      for (int ni = 0; ni < 2; ++ni)
        acc[mi][ni] = __builtin_amdgcn_wmma_f32_16x16x32_bf16(
            false, afrag[mi], false, bfrag[ni], (short)0, acc[mi][ni], false, false);

    __syncthreads();    // all LDS reads of tile i done before buffer is re-filled
  }

  // epilogue: C/D layout — VGPR r: lanes0-15 M=r, lanes16-31 M=8+r; N=l16
  #pragma unroll
  for (int ni = 0; ni < 2; ++ni) {
    int n = gN + wn * 32 + ni * 16 + l16;
    float bb = bias[n];
    #pragma unroll
    for (int mi = 0; mi < 4; ++mi) {
      int mbase = gM + wm * 64 + mi * 16 + half * 8;
      #pragma unroll
      for (int r = 0; r < 8; ++r)
        C[(size_t)(mbase + r) * Vv + n] = acc[mi][ni][r] + bb;
    }
  }
}

extern "C" void kernel_launch(void* const* d_in, const int* in_sizes, int n_in,
                              void* d_out, int out_size, void* d_ws, size_t ws_size,
                              hipStream_t stream) {
  const int*   x     = (const int*)  d_in[0];
  const float* h0    = (const float*)d_in[1];
  const float* emb   = (const float*)d_in[2];
  const float* U_w   = (const float*)d_in[3];
  const float* out_w = (const float*)d_in[4];
  const float* out_b = (const float*)d_in[5];
  float* logits = (float*)d_out;

  uint8_t* ws = (uint8_t*)d_ws;
  float*    e   = (float*)ws;                                  // 8 MB gathered embeddings
  float*    hA  = (float*)(ws + (size_t)(8u << 20));           // ping
  float*    hB  = hA + Bb * Dd;                                // pong
  unsigned* cnt = (unsigned*)(ws + (size_t)(8u << 20) + 65536);// barrier counter
  bf16_t*   hs  = (bf16_t*)(ws + (size_t)(9u << 20));          // 4 MB bf16 activations
  bf16_t*   wb  = (bf16_t*)(ws + (size_t)(13u << 20));         // 64 MB bf16 weights

  atlm_init  <<<16, 256, 0, stream>>>(h0, hA, cnt);
  atlm_gather<<<Mm, 256, 0, stream>>>(x, emb, e);
  atlm_cvt_w <<<(Vv * Dd) / 1024, 256, 0, stream>>>(out_w, wb);
  atlm_rnn   <<<128, 256, 0, stream>>>(e, U_w, hA, hB, hs, cnt);
  atlm_gemm  <<<dim3(Vv / BN, Mm / BM), 256, 0, stream>>>(hs, wb, out_b, logits);
}
